// GATBase_6433861009603
// MI455X (gfx1250) — matrix operand order repaired and checked by reading, not compiled, not run on hardware
//
#include <hip/hip_runtime.h>
#include <hip/hip_bf16.h>
#include <stdint.h>

// ---------------- CDNA5 WMMA types ----------------
typedef __attribute__((ext_vector_type(16))) __bf16 v16bf;
typedef __attribute__((ext_vector_type(8)))  float  v8f;

#define NEG_SLOPE 0.2f

// ---------------- helpers ----------------
static __device__ __forceinline__ unsigned short f32_to_bf16_rne(float f) {
  unsigned int u = __float_as_uint(f);
  unsigned int r = u + 0x7FFFu + ((u >> 16) & 1u);
  return (unsigned short)(r >> 16);
}

static __device__ __forceinline__ void atomicMaxF(float* addr, float v) {
  // classic sign-split trick: int-max for v>=0, uint-min for v<0
  if (v >= 0.0f) atomicMax((int*)addr, __float_as_int(v));
  else           atomicMin((unsigned int*)addr, __float_as_uint(v));
}

// ---------------- utility kernels ----------------
__global__ void k_fill(float* __restrict__ p, int n, float v) {
  int i = blockIdx.x * 256 + threadIdx.x;
  if (i < n) p[i] = v;
}

__global__ void k_cvt_bf16(const float* __restrict__ in, unsigned short* __restrict__ out, int n) {
  int i = blockIdx.x * 256 + threadIdx.x;
  if (i < n) out[i] = f32_to_bf16_rne(in[i]);
}

// Pack 128x128 row-major f32 W into WMMA B fragments, contiguous k order matching the A loads (bf16).
// Fragment f = (K/32)*8 + (n/16); within a fragment: lane = ((K/16)&1)*16 + (n&15),
// 16 bf16 per lane: dword j = (K&15)/2, half = K&1.  Fragment = 512 ushorts.
__global__ void k_pack_b(const float* __restrict__ B, unsigned short* __restrict__ Bp) {
  int t = blockIdx.x * 256 + threadIdx.x;
  if (t >= 128 * 128) return;
  int K = t >> 7, n = t & 127;
  int f    = ((K >> 5) << 3) | (n >> 4);
  int lane = (((K >> 4) & 1) << 4) | (n & 15);
  int j    = (K & 15) >> 1;
  int half = K & 1;
  Bp[f * 512 + lane * 16 + j * 2 + half] = f32_to_bf16_rne(B[t]);
}

union FragU { uint4 u[2]; v16bf v; };

// ---------------- bf16 WMMA GEMM:  C[M,128] = A[M,128] * B[128,128] ----------------
// 256 threads = 8 waves; wave w computes rows [blk*128 + 16w, +16), all 128 cols.
// B (packed fragments, 32KB) is staged once per workgroup into LDS via the CDNA5
// async-copy path (ASYNCcnt), then the K-loop feeds WMMA from ds_load_b128.
__global__ __launch_bounds__(256) void k_gemm_bf16(const unsigned short* __restrict__ A,
                                                   const unsigned short* __restrict__ Bp,
                                                   float* __restrict__ C, int M) {
  __shared__ unsigned short sB[128 * 128];   // 32 KB packed B fragments

  // --- async stage B -> LDS: 256 lanes x 8 x 16B = 32KB ---
  {
    const int tid = threadIdx.x;
    unsigned long long gbase = (unsigned long long)(const void*)Bp;
    unsigned lbase = (unsigned)(uintptr_t)(&sB[0]);
    #pragma unroll
    for (int i = 0; i < 8; ++i) {
      unsigned byteoff = (unsigned)(tid + i * 256) * 16u;
      unsigned lds_addr = lbase + byteoff;
      unsigned long long ga = gbase + byteoff;
      asm volatile("global_load_async_to_lds_b128 %0, %1, off"
                   :: "v"(lds_addr), "v"(ga) : "memory");
    }
    asm volatile("s_wait_asynccnt 0" ::: "memory");
  }
  __syncthreads();

  const int lane = threadIdx.x & 31;
  const int wave = threadIdx.x >> 5;
  const int rowBase = blockIdx.x * 128 + wave * 16;
  const int g = lane >> 4;
  int row = rowBase + (lane & 15);
  if (row > M - 1) row = M - 1;                 // clamp OOB rows (stores are guarded)
  const unsigned short* arow = A + (long)row * 128;
  __builtin_prefetch(arow + 128 * 16, 0, 1);    // hint next tile's row into cache

  v8f acc[8] = {};
  #pragma unroll
  for (int kb = 0; kb < 128; kb += 32) {
    FragU a;
    a.u[0] = *(const uint4*)(arow + kb + g * 16);
    a.u[1] = *(const uint4*)(arow + kb + g * 16 + 8);
    const unsigned short* bb = sB + ((kb >> 5) << 3) * 512 + lane * 16;
    #pragma unroll
    for (int nb = 0; nb < 8; ++nb) {
      FragU b;
      b.u[0] = *(const uint4*)(bb + nb * 512);      // ds_load_b128
      b.u[1] = *(const uint4*)(bb + nb * 512 + 8);  // ds_load_b128
      acc[nb] = __builtin_amdgcn_wmma_f32_16x16x32_bf16(
          false, a.v, false, b.v, (short)0, acc[nb], false, false);
    }
  }

  // D layout: lanes 0-15 hold M=r (VGPR r), lanes 16-31 hold M=8+r; N = lane&15
  const bool fullTile = (rowBase + 16 <= M);
  #pragma unroll
  for (int nb = 0; nb < 8; ++nb) {
    int col = (nb << 4) + (lane & 15);
    if (fullTile) {
      float* cbase = C + (long)(rowBase + (g << 3)) * 128 + col;
      #pragma unroll
      for (int r = 0; r < 8; ++r) cbase[(long)r * 128] = acc[nb][r];
    } else {
      #pragma unroll
      for (int r = 0; r < 8; ++r) {
        int rr = rowBase + r + (g << 3);
        if (rr < M) C[(long)rr * 128 + col] = acc[nb][r];
      }
    }
  }
}

// ---------------- attention scores: s[n,h] = dot(xp[n, h*C:...], a[h]) ----------------
__global__ void k_scores(const float* __restrict__ xp,
                         const float* __restrict__ asrc, const float* __restrict__ adst,
                         float* __restrict__ ssrc, float* __restrict__ sdst,
                         int total, int hmask, int C) {
  int t = blockIdx.x * 256 + threadIdx.x;
  if (t >= total) return;
  int h = t & hmask;
  int n = t / (hmask + 1);
  const float* v = xp + (long)n * 128 + h * C;
  float ss = 0.f, sd = 0.f;
  for (int c = 0; c < C; ++c) {
    float x = v[c];
    ss += x * asrc[h * C + c];
    sd += x * adst[h * C + c];
  }
  ssrc[t] = ss; sdst[t] = sd;
}

// ---------------- edge passes (edges 0..E-1 real, E..E+N-1 self loops) ----------------
static __device__ __forceinline__ void edge_decode(const int* ei, int e, int E, int& src, int& dst) {
  if (e < E) { src = ei[e]; dst = ei[E + e]; }
  else       { src = dst = e - E; }
}

__global__ void k_edge_max(const int* __restrict__ ei,
                           const float* __restrict__ ssrc, const float* __restrict__ sdst,
                           float* __restrict__ m, int E, int N, int hshift, int hmask) {
  int t = blockIdx.x * 256 + threadIdx.x;
  int tot = (E + N) << hshift;
  if (t >= tot) return;
  int h = t & hmask, e = t >> hshift;
  int src, dst; edge_decode(ei, e, E, src, dst);
  float v = ssrc[(src << hshift) + h] + sdst[(dst << hshift) + h];
  v = v > 0.f ? v : NEG_SLOPE * v;
  atomicMaxF(&m[(dst << hshift) + h], v);
}

__global__ void k_edge_sum(const int* __restrict__ ei,
                           const float* __restrict__ ssrc, const float* __restrict__ sdst,
                           const float* __restrict__ m, float* __restrict__ denom,
                           int E, int N, int hshift, int hmask) {
  int t = blockIdx.x * 256 + threadIdx.x;
  int tot = (E + N) << hshift;
  if (t >= tot) return;
  int h = t & hmask, e = t >> hshift;
  int src, dst; edge_decode(ei, e, E, src, dst);
  float v = ssrc[(src << hshift) + h] + sdst[(dst << hshift) + h];
  v = v > 0.f ? v : NEG_SLOPE * v;
  float ex = __expf(v - m[(dst << hshift) + h]);
  atomicAdd(&denom[(dst << hshift) + h], ex);
}

__global__ void k_edge_scatter(const int* __restrict__ ei,
                               const float* __restrict__ ssrc, const float* __restrict__ sdst,
                               const float* __restrict__ m, const float* __restrict__ denom,
                               const float* __restrict__ xp, float* __restrict__ acc,
                               int E, int N, int hshift, int hmask, int C) {
  int t = blockIdx.x * 256 + threadIdx.x;
  int tot = (E + N) << hshift;
  if (t >= tot) return;
  int h = t & hmask, e = t >> hshift;
  int src, dst; edge_decode(ei, e, E, src, dst);
  float v = ssrc[(src << hshift) + h] + sdst[(dst << hshift) + h];
  v = v > 0.f ? v : NEG_SLOPE * v;
  int di = (dst << hshift) + h;
  float alpha = __expf(v - m[di]) / (denom[di] + 1e-16f);
  const float* xs = xp + (long)src * 128 + h * C;
  float* ad = acc + (long)dst * 128 + h * C;
  for (int c = 0; c < C; c += 4) {
    float4 q = *(const float4*)(xs + c);
    atomicAdd(ad + c + 0, q.x * alpha);
    atomicAdd(ad + c + 1, q.y * alpha);
    atomicAdd(ad + c + 2, q.z * alpha);
    atomicAdd(ad + c + 3, q.w * alpha);
  }
}

// ---------------- bias / activation ----------------
__global__ void k_elu_bias(float* __restrict__ a, const float* __restrict__ b, int n) {
  int i = blockIdx.x * 256 + threadIdx.x;
  if (i < n) {
    float v = a[i] + b[i & 127];
    a[i] = v > 0.f ? v : (__expf(v) - 1.0f);
  }
}

__global__ void k_add_bias(float* __restrict__ a, const float* __restrict__ b, int n) {
  int i = blockIdx.x * 256 + threadIdx.x;
  if (i < n) a[i] += b[i & 127];
}

// ---------------- launcher ----------------
extern "C" void kernel_launch(void* const* d_in, const int* in_sizes, int n_in,
                              void* d_out, int out_size, void* d_ws, size_t ws_size,
                              hipStream_t stream) {
  const float* x   = (const float*)d_in[0];
  const int*   ei  = (const int*)  d_in[1];
  const float* W1  = (const float*)d_in[2];
  const float* a1s = (const float*)d_in[3];
  const float* a1d = (const float*)d_in[4];
  const float* b1  = (const float*)d_in[5];
  const float* W2  = (const float*)d_in[6];
  const float* a2s = (const float*)d_in[7];
  const float* a2d = (const float*)d_in[8];
  const float* b2  = (const float*)d_in[9];
  float* out = (float*)d_out;

  const int N = in_sizes[0] / 128;
  const int E = in_sizes[1] / 2;
  const int H = 4;

  // workspace layout
  char* ws = (char*)d_ws;
  size_t off = 0;
  auto alloc = [&](size_t bytes) -> void* {
    void* p = ws + off;
    off += (bytes + 255) & ~(size_t)255;
    return p;
  };
  float*          xp   = (float*)alloc((size_t)N * 128 * 4);   // projected features
  float*          acc  = (float*)alloc((size_t)N * 128 * 4);   // aggregation / h
  unsigned short* abf  = (unsigned short*)alloc((size_t)N * 128 * 2);
  unsigned short* wpk  = (unsigned short*)alloc(128 * 128 * 2);
  float*          ssrc = (float*)alloc((size_t)N * H * 4);
  float*          sdst = (float*)alloc((size_t)N * H * 4);
  float*          mbuf = (float*)alloc((size_t)N * H * 4);
  float*          dbuf = (float*)alloc((size_t)N * H * 4);

  const int nEl = N * 128;
  const int blkEl = (nEl + 255) / 256;
  const int gemmBlocks = (N + 127) / 128;
  const float NEG_INF = -__builtin_inff();

  // ================= layer 1 (H=4, C=32) =================
  k_cvt_bf16<<<blkEl, 256, 0, stream>>>(x, abf, nEl);
  k_pack_b<<<64, 256, 0, stream>>>(W1, wpk);
  k_gemm_bf16<<<gemmBlocks, 256, 0, stream>>>(abf, wpk, xp, N);

  {
    int tot = N * H;
    k_scores<<<(tot + 255) / 256, 256, 0, stream>>>(xp, a1s, a1d, ssrc, sdst, tot, H - 1, 32);
    k_fill<<<(tot + 255) / 256, 256, 0, stream>>>(mbuf, tot, NEG_INF);
    k_fill<<<(tot + 255) / 256, 256, 0, stream>>>(dbuf, tot, 0.0f);
    k_fill<<<blkEl, 256, 0, stream>>>(acc, nEl, 0.0f);
    int etot = (E + N) * H;
    int eblk = (etot + 255) / 256;
    k_edge_max<<<eblk, 256, 0, stream>>>(ei, ssrc, sdst, mbuf, E, N, 2, 3);
    k_edge_sum<<<eblk, 256, 0, stream>>>(ei, ssrc, sdst, mbuf, dbuf, E, N, 2, 3);
    k_edge_scatter<<<eblk, 256, 0, stream>>>(ei, ssrc, sdst, mbuf, dbuf, xp, acc, E, N, 2, 3, 32);
    k_elu_bias<<<blkEl, 256, 0, stream>>>(acc, b1, nEl);   // acc = h = elu(agg + b1)
  }

  // ================= layer 2 (H=1, C=128) =================
  k_cvt_bf16<<<blkEl, 256, 0, stream>>>(acc, abf, nEl);
  k_pack_b<<<64, 256, 0, stream>>>(W2, wpk);
  k_gemm_bf16<<<gemmBlocks, 256, 0, stream>>>(abf, wpk, xp, N);

  {
    int tot = N;
    k_scores<<<(tot + 255) / 256, 256, 0, stream>>>(xp, a2s, a2d, ssrc, sdst, tot, 0, 128);
    k_fill<<<(tot + 255) / 256, 256, 0, stream>>>(mbuf, tot, NEG_INF);
    k_fill<<<(tot + 255) / 256, 256, 0, stream>>>(dbuf, tot, 0.0f);
    k_fill<<<blkEl, 256, 0, stream>>>(out, nEl, 0.0f);
    int etot = (E + N);
    int eblk = (etot + 255) / 256;
    k_edge_max<<<eblk, 256, 0, stream>>>(ei, ssrc, sdst, mbuf, E, N, 0, 0);
    k_edge_sum<<<eblk, 256, 0, stream>>>(ei, ssrc, sdst, mbuf, dbuf, E, N, 0, 0);
    k_edge_scatter<<<eblk, 256, 0, stream>>>(ei, ssrc, sdst, mbuf, dbuf, xp, out, E, N, 0, 0, 128);
    k_add_bias<<<blkEl, 256, 0, stream>>>(out, b2, nEl);
  }
}
